// GCN_20186346291924
// MI455X (gfx1250) — matrix operand order repaired and checked by reading, not compiled, not run on hardware
//
#include <hip/hip_runtime.h>

typedef __attribute__((ext_vector_type(16))) __bf16 v16bf;
typedef __attribute__((ext_vector_type(8)))  __bf16 v8bf;
typedef __attribute__((ext_vector_type(4)))  __bf16 v4bf;
typedef __attribute__((ext_vector_type(8)))  float  v8f;
typedef __attribute__((ext_vector_type(4)))  float  v4f;

#define BATCH 256
#define NNODE 512
#define DDIM  128
#define FDIM  16

// Padded LDS row strides (bf16 elems): stride%64dw == 4 -> conflict-free row fans
#define XS 136   // x   [512][XS]
#define MS 520   // m^T [128][MS]
#define WS 136   // W^T [128][WS], Wout^T [16][WS]

__device__ __forceinline__ v8f wmma_bf16(v16bf a, v16bf b, v8f c) {
  // D = A(16x32 bf16) * B(32x16 bf16) + C(16x16 f32)
  return __builtin_amdgcn_wmma_f32_16x16x32_bf16(false, a, false, b, (short)0, c,
                                                 false, false);
}

__device__ __forceinline__ v8f zero8() {
  v8f z;
#pragma unroll
  for (int i = 0; i < 8; ++i) z[i] = 0.0f;
  return z;
}

// Load 16 bf16 as two 16B chunks (chunk at p, chunk at p+gap elems)
__device__ __forceinline__ v16bf ld16(const __bf16* p, int gap) {
  v8bf lo = *(const v8bf*)p;
  v8bf hi = *(const v8bf*)(p + gap);
  return __builtin_shufflevector(lo, hi, 0, 1, 2, 3, 4, 5, 6, 7,
                                 8, 9, 10, 11, 12, 13, 14, 15);
}

__device__ __forceinline__ v16bf cvtA(v4f q0, v4f q1, v4f q2, v4f q3) {
  v16bf A;
  A[0]  = (__bf16)q0[0]; A[1]  = (__bf16)q0[1]; A[2]  = (__bf16)q0[2]; A[3]  = (__bf16)q0[3];
  A[4]  = (__bf16)q1[0]; A[5]  = (__bf16)q1[1]; A[6]  = (__bf16)q1[2]; A[7]  = (__bf16)q1[3];
  A[8]  = (__bf16)q2[0]; A[9]  = (__bf16)q2[1]; A[10] = (__bf16)q2[2]; A[11] = (__bf16)q2[3];
  A[12] = (__bf16)q3[0]; A[13] = (__bf16)q3[1]; A[14] = (__bf16)q3[2]; A[15] = (__bf16)q3[3];
  return A;
}

__global__ __launch_bounds__(512, 1)
void gcn_fused_kernel(const float* __restrict__ h, const float* __restrict__ adj,
                      const float* __restrict__ nmask,
                      const float* __restrict__ W1, const float* __restrict__ b1,
                      const float* __restrict__ W2, const float* __restrict__ b2,
                      const float* __restrict__ W3, const float* __restrict__ b3,
                      const float* __restrict__ Wo, const float* __restrict__ bo,
                      float* __restrict__ out)
{
  __shared__ __attribute__((aligned(32))) __bf16 sXA[NNODE * XS];  // x, bf16 row-major
  __shared__ __attribute__((aligned(32))) __bf16 sMT[DDIM * MS];   // m^T, bf16
  __shared__ __attribute__((aligned(32))) __bf16 sWT[DDIM * WS];   // W^T
  __shared__ __attribute__((aligned(32))) __bf16 sWoT[FDIM * WS];  // W_out^T
  __shared__ float sBias[DDIM];
  __shared__ float sBout[FDIM];
  __shared__ float sMask[NNODE];

  const int tid  = threadIdx.x;        // 0..511 (16 waves of 32)
  const int b    = blockIdx.x;         // batch element
  const int lane = tid & 31;
  const int wv   = tid >> 5;           // wave id 0..15
  const int l16  = lane & 15;
  const int half = lane >> 4;          // K-half selector for WMMA operand layout

  // ---------------- Phase 0: stage h[b] (f32 -> bf16), mask, W_out^T, b_out ----
  {
    const float* hb = h + (size_t)b * NNODE * DDIM;
#pragma unroll 1
    for (int i = 0; i < 32; ++i) {
      int idx4 = tid + i * 512;              // 0..16383 float4s
      int row  = idx4 >> 5;                  // 32 float4 per 128-wide row
      int col  = (idx4 & 31) << 2;
      v4f q = *(const v4f*)(hb + ((size_t)idx4 << 2));
      v4bf p;
      p[0] = (__bf16)q[0]; p[1] = (__bf16)q[1];
      p[2] = (__bf16)q[2]; p[3] = (__bf16)q[3];
      *(v4bf*)(&sXA[row * XS + col]) = p;
    }
    sMask[tid] = nmask[(size_t)b * NNODE + tid];
#pragma unroll
    for (int i = 0; i < 4; ++i) {            // W_out: 128x16 -> WoT[16][128]
      int e = tid + i * 512;                 // 0..2047
      int k = e >> 4, n = e & 15;
      sWoT[n * WS + k] = (__bf16)Wo[e];
    }
    if (tid < FDIM) sBout[tid] = bo[tid];
  }

  const float* adjb = adj + (size_t)b * NNODE * NNODE;

  // ------- 3 GC layers: layer loop rolled (code size), stripes UNROLLED -------
  // (stripe loops must stay unrolled: rolling them makes the W^T B-tiles
  //  loop-invariant -> LICM hoists 512B of WMMA operands into scratch spills)
#pragma unroll 1
  for (int l = 0; l < 3; ++l) {
    const float* W  = (l == 0) ? W1 : ((l == 1) ? W2 : W3);
    const float* bb = (l == 0) ? b1 : ((l == 1) ? b2 : b3);

    // stage W^T + bias
#pragma unroll 1
    for (int i = 0; i < 32; ++i) {
      int idx4 = tid + i * 512;              // 16384 elems as float4
      int k  = idx4 >> 5;
      int n4 = (idx4 & 31) << 2;
      v4f q = *(const v4f*)(W + ((size_t)idx4 << 2));
      sWT[(n4 + 0) * WS + k] = (__bf16)q[0];
      sWT[(n4 + 1) * WS + k] = (__bf16)q[1];
      sWT[(n4 + 2) * WS + k] = (__bf16)q[2];
      sWT[(n4 + 3) * WS + k] = (__bf16)q[3];
    }
    if (tid < DDIM) sBias[tid] = bb[tid];
    __syncthreads();

    // -------- transform: m = x @ W + b  ->  sMT (stored transposed) --------
#pragma unroll
    for (int st = 0; st < 2; ++st) {
      const int m0 = (wv + st * 16) * 16;    // node-row stripe
      v8f acc[8];
#pragma unroll
      for (int n = 0; n < 8; ++n) acc[n] = zero8();
#pragma unroll
      for (int kk = 0; kk < 4; ++kk) {
        const int k0 = kk * 32;
        v16bf A  = ld16(&sXA[(m0 + l16) * XS + k0 + half * 8], 16);
        v16bf Bc = ld16(&sWT[l16 * WS + k0 + half * 8], 16);   // n = 0
#pragma unroll
        for (int n = 0; n < 8; ++n) {
          v16bf Bn = Bc;
          if (n < 7)                                            // issue B[n+1] early
            Bn = ld16(&sWT[((n + 1) * 16 + l16) * WS + k0 + half * 8], 16);
          acc[n] = wmma_bf16(A, Bc, acc[n]);
          Bc = Bn;
        }
      }
#pragma unroll
      for (int n = 0; n < 8; ++n) {
        const int col = n * 16 + l16;        // output feature
        const float bv = sBias[col];
        v8bf p;
#pragma unroll
        for (int r = 0; r < 8; ++r) p[r] = (__bf16)(acc[n][r] + bv);
        // D layout: lane holds N=col, M = m0 + half*8 + r  -> contiguous in m^T
        *(v8bf*)(&sMT[col * MS + m0 + half * 8]) = p;
      }
    }
    __syncthreads();

    // -------- aggregate: x = relu(adj @ m) -> sXA (row-major bf16) --------
#pragma unroll
    for (int st = 0; st < 2; ++st) {
      const int m0 = (wv + st * 16) * 16;
      const float* arow = adjb + (size_t)(m0 + l16) * NNODE; // adj row for this lane
      v8f acc[8];
#pragma unroll
      for (int n = 0; n < 8; ++n) acc[n] = zero8();

      // depth-1 register pipeline on the adj stream
      const int c0 = half * 8;
      v4f q0 = *(const v4f*)(arow + c0);
      v4f q1 = *(const v4f*)(arow + c0 + 4);
      v4f q2 = *(const v4f*)(arow + c0 + 16);
      v4f q3 = *(const v4f*)(arow + c0 + 20);

#pragma unroll 1
      for (int j0 = 0; j0 < NNODE; j0 += 32) {
        // pull the line 3 K-steps ahead into cache (1 line per row per step)
        if (j0 + 96 < NNODE)
          __builtin_prefetch((const void*)(arow + j0 + 96), 0, 1);

        v4f p0 = q0, p1 = q1, p2 = q2, p3 = q3;
        const int jn = j0 + 32;
        if (jn < NNODE) {                     // issue next K-step's loads now
          const int cN = jn + half * 8;
          p0 = *(const v4f*)(arow + cN);
          p1 = *(const v4f*)(arow + cN + 4);
          p2 = *(const v4f*)(arow + cN + 16);
          p3 = *(const v4f*)(arow + cN + 20);
        }

        v16bf A  = cvtA(q0, q1, q2, q3);
        v16bf Bc = ld16(&sMT[l16 * MS + j0 + half * 8], 16);   // n = 0
#pragma unroll
        for (int n = 0; n < 8; ++n) {
          v16bf Bn = Bc;
          if (n < 7)
            Bn = ld16(&sMT[((n + 1) * 16 + l16) * MS + j0 + half * 8], 16);
          acc[n] = wmma_bf16(A, Bc, acc[n]);
          Bc = Bn;
        }
        q0 = p0; q1 = p1; q2 = p2; q3 = p3;
      }

#pragma unroll
      for (int n = 0; n < 8; ++n) {
        const int col = n * 16 + l16;
#pragma unroll
        for (int r = 0; r < 8; ++r) {
          float v = acc[n][r];
          v = v > 0.0f ? v : 0.0f;           // ReLU (NORM == 1.0)
          sXA[(m0 + half * 8 + r) * XS + col] = (__bf16)v;
        }
      }
    }
    __syncthreads();
  }

  // ---------------- output: (x @ W_out + b_out) * node_mask ----------------
#pragma unroll
  for (int st = 0; st < 2; ++st) {
    const int m0 = (wv + st * 16) * 16;
    v8f acc = zero8();
#pragma unroll
    for (int kk = 0; kk < 4; ++kk) {
      const int k0 = kk * 32;
      v16bf A = ld16(&sXA[(m0 + l16) * XS + k0 + half * 8], 16);
      v16bf B = ld16(&sWoT[l16 * WS + k0 + half * 8], 16);
      acc = wmma_bf16(A, B, acc);
    }
    const int col = l16;                     // F = 16 -> single N tile
    const float bv = sBout[col];
    float* ob = out + ((size_t)b * NNODE + m0 + half * 8) * FDIM + col;
#pragma unroll
    for (int r = 0; r < 8; ++r) {
      const int m = m0 + half * 8 + r;
      ob[(size_t)r * FDIM] = (acc[r] + bv) * sMask[m];
    }
  }
}

extern "C" void kernel_launch(void* const* d_in, const int* in_sizes, int n_in,
                              void* d_out, int out_size, void* d_ws, size_t ws_size,
                              hipStream_t stream) {
  (void)in_sizes; (void)n_in; (void)out_size; (void)d_ws; (void)ws_size;
  const float* h  = (const float*)d_in[0];
  const float* a  = (const float*)d_in[1];
  const float* nm = (const float*)d_in[2];
  const float* W1 = (const float*)d_in[3];
  const float* b1 = (const float*)d_in[4];
  const float* W2 = (const float*)d_in[5];
  const float* b2 = (const float*)d_in[6];
  const float* W3 = (const float*)d_in[7];
  const float* b3 = (const float*)d_in[8];
  const float* Wo = (const float*)d_in[9];
  const float* bo = (const float*)d_in[10];
  hipLaunchKernelGGL(gcn_fused_kernel, dim3(BATCH), dim3(512), 0, stream,
                     h, a, nm, W1, b1, W2, b2, W3, b3, Wo, bo, (float*)d_out);
}
